// LSTMDecoder_85169201480407
// MI455X (gfx1250) — matrix operand order repaired and checked by reading, not compiled, not run on hardware
//
#include <hip/hip_runtime.h>

#define HID     256
#define HOR     24
#define BATCHN  32768
#define MROWS   32            // batch rows per workgroup
#define LSTRIDE 264           // bf16 elems per LDS h row (256 + 8 pad -> 528B, 16B aligned)

typedef __bf16 v16bf __attribute__((ext_vector_type(16)));
typedef __bf16 v8bf  __attribute__((ext_vector_type(8)));
typedef float  v8f   __attribute__((ext_vector_type(8)));

__device__ __forceinline__ float fast_sigmoid(float v) {
    return __builtin_amdgcn_rcpf(1.0f + __expf(-v));
}
__device__ __forceinline__ float fast_tanh(float v) {
    return 2.0f * __builtin_amdgcn_rcpf(1.0f + __expf(-2.0f * v)) - 1.0f;
}

// One-time f32 -> bf16 conversion of W_hh (B operand, L2-resident afterwards).
__global__ void __launch_bounds__(256)
pack_w_kernel(const float* __restrict__ w, __bf16* __restrict__ out, int n) {
    int i = blockIdx.x * 256 + threadIdx.x;
    if (i < n) out[i] = (__bf16)w[i];
}

__global__ void __launch_bounds__(256)
lstm_kernel(const float* __restrict__ x, const float* __restrict__ h0,
            const float* __restrict__ c0, const float* __restrict__ w_ih,
            const __bf16* __restrict__ wbf, const float* __restrict__ b_ih,
            const float* __restrict__ b_hh, const float* __restrict__ fc_w,
            const float* __restrict__ fc_b, float* __restrict__ out)
{
    __shared__ __bf16 ldsH[MROWS * LSTRIDE];   // h_t, bf16, row-major [32][256+pad]
    __shared__ float  part[8][MROWS];          // per-wave fc partial sums

    const int tid  = threadIdx.x;
    const int wave = tid >> 5;
    const int lane = tid & 31;
    const int l15  = lane & 15;
    const int lh   = lane >> 4;
    const int r0   = blockIdx.x * MROWS;

    // ---- stage initial h (f32 -> bf16) into LDS ----
    {
        const int row = tid >> 3;            // 0..31
        const int kb  = (tid & 7) * 32;      // 0..224
        const float* src = h0 + (size_t)(r0 + row) * HID + kb;
        __bf16* dst = ldsH + row * LSTRIDE + kb;
        #pragma unroll
        for (int k = 0; k < 32; ++k) dst[k] = (__bf16)src[k];
    }

    // ---- persistent per-lane state ----
    // c in registers, C-matrix layout: (ct,mt,r) -> row = mt*16+r+8*lh, col = 32*wave+16*ct+l15
    float cr[2][2][8];
    #pragma unroll
    for (int ct = 0; ct < 2; ++ct)
        #pragma unroll
        for (int mt = 0; mt < 2; ++mt)
            #pragma unroll
            for (int r = 0; r < 8; ++r)
                cr[ct][mt][r] =
                    c0[(size_t)(r0 + mt * 16 + r + 8 * lh) * HID + 32 * wave + 16 * ct + l15];

    // x fragment in C layout (x never changes in the reference loop)
    float xf[2][8];
    #pragma unroll
    for (int mt = 0; mt < 2; ++mt)
        #pragma unroll
        for (int r = 0; r < 8; ++r)
            xf[mt][r] = x[r0 + mt * 16 + r + 8 * lh];

    // per-gate column constants + B-fragment base offsets
    // wave w owns n-tiles {2w+ct + 16*g} -> i/f/g/o columns for hidden units [32w, 32w+32)
    float wv[4][2], bv[4][2];
    int   boff[4][2];
    #pragma unroll
    for (int g = 0; g < 4; ++g)
        #pragma unroll
        for (int ct = 0; ct < 2; ++ct) {
            const int ncol = 256 * g + 32 * wave + 16 * ct + l15;
            wv[g][ct]   = w_ih[ncol];
            bv[g][ct]   = b_ih[ncol] + b_hh[ncol];
            boff[g][ct] = ncol * HID + lh * 16;   // bf16-element offset into wbf
        }

    const float fcw0 = fc_w[32 * wave + l15];
    const float fcw1 = fc_w[32 * wave + 16 + l15];
    const float fcb  = fc_b[0];

    // A-fragment base offsets (bf16 elems): lane = row (mt*16+l15), one contiguous 16-k run per tile (k = 16*lh + s)
    const int aoff0 = l15 * LSTRIDE + lh * 16;
    const int aoff1 = aoff0 + 16 * LSTRIDE;

    int inv = 0;   // uniform, laundered each step: defeats LICM without losing
                   // the global address-space provenance of wbf (keeps global_load)

    __syncthreads();

    #pragma unroll 1
    for (int step = 0; step < HOR; ++step) {
        // ---- defeat loop-invariant hoisting (it was spilling ~650 VGPRs to scratch):
        // make the weight offset and the x-fragment regs appear loop-variant, so
        // B-fragment loads stay L2 loads inside the loop and the gate-bias init FMAs
        // are recomputed (cheap) instead of being materialized+spilled.
        asm volatile("" : "+s"(inv));
        #pragma unroll
        for (int mt = 0; mt < 2; ++mt)
            #pragma unroll
            for (int r = 0; r < 8; ++r)
                asm volatile("" : "+v"(xf[mt][r]));

        const __bf16* wp = wbf + inv;

        float hn[2][2][8];   // new h values, carried past the barrier

        // two column passes: keeps only 64 acc VGPRs live at a time
        #pragma unroll
        for (int ct = 0; ct < 2; ++ct) {
            // ---- init accumulators: gates = x*W_ih + (b_ih + b_hh) ----
            v8f acc[4][2];
            #pragma unroll
            for (int g = 0; g < 4; ++g)
                #pragma unroll
                for (int mt = 0; mt < 2; ++mt)
                    #pragma unroll
                    for (int r = 0; r < 8; ++r)
                        acc[g][mt][r] = xf[mt][r] * wv[g][ct] + bv[g][ct];

            // ---- gates += h @ W_hh^T : bf16 WMMA, K=256 in 8 slices ----
            // explicit 1-deep double buffer on the 4 B fragments: kt+1's weight
            // group is in flight while kt's 8 WMMAs execute, so the loadcnt wait
            // in front of each WMMA group has a whole K-slice of slack.
            v16bf bcur[4], bnxt[4];
            #pragma unroll
            for (int g = 0; g < 4; ++g)
                bcur[g] = *(const v16bf*)(wp + boff[g][ct]);

            #pragma unroll
            for (int kt = 0; kt < 8; ++kt) {
                if (kt < 7) {
                    #pragma unroll
                    for (int g = 0; g < 4; ++g)
                        bnxt[g] = *(const v16bf*)(wp + boff[g][ct] + (kt + 1) * 32);
                }
                const v8bf a0lo = *(const v8bf*)(ldsH + aoff0 + kt * 32);
                const v8bf a0hi = *(const v8bf*)(ldsH + aoff0 + kt * 32 + 8);
                const v8bf a1lo = *(const v8bf*)(ldsH + aoff1 + kt * 32);
                const v8bf a1hi = *(const v8bf*)(ldsH + aoff1 + kt * 32 + 8);
                const v16bf a0 = __builtin_shufflevector(a0lo, a0hi,
                    0, 1, 2, 3, 4, 5, 6, 7, 8, 9, 10, 11, 12, 13, 14, 15);
                const v16bf a1 = __builtin_shufflevector(a1lo, a1hi,
                    0, 1, 2, 3, 4, 5, 6, 7, 8, 9, 10, 11, 12, 13, 14, 15);
                #pragma unroll
                for (int g = 0; g < 4; ++g) {
                    acc[g][0] = __builtin_amdgcn_wmma_f32_16x16x32_bf16(
                        false, a0, false, bcur[g], (short)0, acc[g][0], false, false);
                    acc[g][1] = __builtin_amdgcn_wmma_f32_16x16x32_bf16(
                        false, a1, false, bcur[g], (short)0, acc[g][1], false, false);
                }
                #pragma unroll
                for (int g = 0; g < 4; ++g) bcur[g] = bnxt[g];
            }

            // ---- LSTM cell (registers only; LDS untouched until after barrier) ----
            #pragma unroll
            for (int mt = 0; mt < 2; ++mt)
                #pragma unroll
                for (int r = 0; r < 8; ++r) {
                    const float iv = fast_sigmoid(acc[0][mt][r]);
                    const float fv = fast_sigmoid(acc[1][mt][r]);
                    const float gv = fast_tanh   (acc[2][mt][r]);
                    const float ov = fast_sigmoid(acc[3][mt][r]);
                    const float cn = fv * cr[ct][mt][r] + iv * gv;
                    cr[ct][mt][r]  = cn;
                    hn[ct][mt][r]  = ov * fast_tanh(cn);
                }

            // pass boundary: keep VALU/WMMA of the two passes separate (register
            // pressure), but let VMEM-reads (32) and DS-reads (256) cross so the
            // next pass's weight/A loads start under this pass's elementwise chain.
            __builtin_amdgcn_sched_barrier(32 | 256);
        }

        __syncthreads();   // all waves done reading old h from LDS

        // ---- write new h (bf16) + fc partial sums ----
        #pragma unroll
        for (int mt = 0; mt < 2; ++mt)
            #pragma unroll
            for (int r = 0; r < 8; ++r) {
                const int rowoff = (mt * 16 + r + 8 * lh) * LSTRIDE + 32 * wave + l15;
                ldsH[rowoff]      = (__bf16)hn[0][mt][r];
                ldsH[rowoff + 16] = (__bf16)hn[1][mt][r];
                float s = hn[0][mt][r] * fcw0 + hn[1][mt][r] * fcw1;
                s += __shfl_xor(s, 1);
                s += __shfl_xor(s, 2);
                s += __shfl_xor(s, 4);
                s += __shfl_xor(s, 8);
                if (l15 == 0) part[wave][mt * 16 + r + 8 * lh] = s;
            }

        __syncthreads();   // new h + partials visible

        if (tid < MROWS) {
            float o = fcb;
            #pragma unroll
            for (int w = 0; w < 8; ++w) o += part[w][tid];
            out[(size_t)step * BATCHN + r0 + tid] = o;
        }
    }
}

extern "C" void kernel_launch(void* const* d_in, const int* in_sizes, int n_in,
                              void* d_out, int out_size, void* d_ws, size_t ws_size,
                              hipStream_t stream) {
    (void)in_sizes; (void)n_in; (void)out_size; (void)ws_size;
    const float* x    = (const float*)d_in[0];
    const float* h    = (const float*)d_in[1];
    const float* c    = (const float*)d_in[2];
    const float* w_ih = (const float*)d_in[3];
    const float* w_hh = (const float*)d_in[4];
    const float* b_ih = (const float*)d_in[5];
    const float* b_hh = (const float*)d_in[6];
    const float* fc_w = (const float*)d_in[7];
    const float* fc_b = (const float*)d_in[8];

    __bf16* wbf = (__bf16*)d_ws;              // 4H*H bf16 = 512 KB scratch
    const int nW = 4 * HID * HID;
    pack_w_kernel<<<(nW + 255) / 256, 256, 0, stream>>>(w_hh, wbf, nW);
    lstm_kernel<<<BATCHN / MROWS, 256, 0, stream>>>(x, h, c, w_ih, wbf,
                                                    b_ih, b_hh, fc_w, fc_b,
                                                    (float*)d_out);
}
